// FallDetection1DCNN_360777252970
// MI455X (gfx1250) — matrix operand order repaired and checked by reading, not compiled, not run on hardware
//
#include <hip/hip_runtime.h>
#include <hip/hip_bf16.h>

typedef __attribute__((ext_vector_type(16))) _Float16 v16h;
typedef __attribute__((ext_vector_type(2)))  _Float16 h2;
typedef __attribute__((ext_vector_type(8)))  float    v8f;

#define NTHREADS 256
#define NWAVES   8
#define T_IN     4096
#define S1_STRIDE 10                 // 9 input features padded to 10 halves
#define BUFA_HALF 41184              // s_in extent (>= 4111*10+32), also covers h2p
#define H2P_EXT_HALF 33376           // h2p extent (>= 1039*32+96)
#define BUFB_HALF 33136              // h1p extent (>= 2063*16+64)
#define BUFA_BYTES (BUFA_HALF * 2)
#define BUFB_BYTES (BUFB_HALF * 2)
#define DUMP_H1 33128                // h1p slack word, beyond max read 33071
#define DUMP_H2 33368                // h2p slack word, beyond max read 33343

// A-matrix (16x32 f16) K-base for VGPR v within a half-wave (contiguous k = 16*hi + 2*v, matches B loads)
__device__ __forceinline__ int a_kbase(int v, int hi) {
  return 2 * v + hi * 16;
}

__device__ __forceinline__ v8f wmma_f16(v16h a, v16h b, v8f c) {
  return __builtin_amdgcn_wmma_f32_16x16x32_f16(false, a, false, b, (short)0, c,
                                                false, false);
}

// 16 consecutive f16 from a 4-byte-aligned LDS address (stage-1 stride is odd*4B)
__device__ __forceinline__ v16h ld16h_a4(const _Float16* p) {
  union { v16h h; unsigned int u[8]; } r;
  const unsigned int* q = reinterpret_cast<const unsigned int*>(p);
#pragma unroll
  for (int i = 0; i < 8; ++i) r.u[i] = q[i];
  return r.h;
}

// lane^1 partner's dword via DPP quad_perm(1,0,3,2) -- pure VALU, no LDS pipe
__device__ __forceinline__ unsigned dpp_xor1_u32(unsigned x) {
  return (unsigned)__builtin_amdgcn_update_dpp(0, (int)x, 0xB1, 0xF, 0xF, true);
}

__device__ __forceinline__ float max_xor1_f32(float x) {
  unsigned yi = dpp_xor1_u32(__builtin_bit_cast(unsigned, x));
  return fmaxf(x, __builtin_bit_cast(float, yi));
}

// packed f16 max without the compiler's canonicalization of the DPP operand
__device__ __forceinline__ unsigned pk_max_u(unsigned a, unsigned b) {
  unsigned d;
  asm("v_pk_max_num_f16 %0, %1, %2" : "=v"(d) : "v"(a), "v"(b));
  return d;
}

// relu + pool2 + f16 pack/store (bias already folded into WMMA C operand).
// Lane pair splits the 8 channels (even lane stores ch 0..3, odd lane 4..7).
// Out-of-range p redirects stores to a dump word in LDS slack (never read).
__device__ __forceinline__ void relu_pool_store(
    const v8f& c, _Float16* buf, int stride,
    int p, int plimit, int mbase, int odd, int dump) {
  unsigned pk[4];
#pragma unroll
  for (int i = 0; i < 4; ++i) {
    h2 hv; hv[0] = (_Float16)c[2 * i]; hv[1] = (_Float16)c[2 * i + 1];
    h2 hz; hz[0] = (_Float16)0.f; hz[1] = (_Float16)0.f;
    hv = __builtin_elementwise_max(hv, hz);             // relu (v_pk_max vs 0)
    unsigned u = __builtin_bit_cast(unsigned, hv);
    pk[i] = pk_max_u(u, dpp_xor1_u32(u));               // pool2
  }
  unsigned s0 = odd ? pk[2] : pk[0];                    // flat 2-way selects
  unsigned s1 = odd ? pk[3] : pk[1];
  bool okp = (p < plimit);
  int base = (p + 2) * stride + mbase + odd * 4;
  int b0 = okp ? base : dump;
  int b1 = okp ? (base + 2) : dump;
  *reinterpret_cast<unsigned*>(&buf[b0]) = s0;
  *reinterpret_cast<unsigned*>(&buf[b1]) = s1;
}

__global__ __launch_bounds__(NTHREADS) void fd_cnn_rows(
    const float* __restrict__ x,
    const float* __restrict__ w1, const float* __restrict__ b1,
    const float* __restrict__ w2, const float* __restrict__ b2,
    const float* __restrict__ w3, const float* __restrict__ b3,
    float* __restrict__ feat) {
  __shared__ __align__(32) unsigned char smem[BUFA_BYTES + BUFB_BYTES];
  __shared__ float feat_acc[16];
  _Float16* bufA = (_Float16*)smem;                  // f16 input row, later h2p
  _Float16* h1p  = (_Float16*)(smem + BUFA_BYTES);   // pooled conv1 activations

  const int tid  = threadIdx.x;
  const int lane = tid & 31;
  const int wave = tid >> 5;
  const int n    = lane & 15;   // N (time) column of C tile; also M row of A
  const int hi   = lane >> 4;   // half-wave select
  const int odd  = lane & 1;    // which channel-quad this lane stores
  const int row  = blockIdx.x;  // b*8 + j

  // ---- zero all LDS (covers halo rows, pad feature column, read slack) ----
  {
    unsigned int* z = (unsigned int*)smem;
    const int nz = (BUFA_BYTES + BUFB_BYTES) / 4;
    for (int i = tid; i < nz; i += NTHREADS) z[i] = 0u;
    if (tid < 16) feat_acc[tid] = 0.f;
  }
  __syncthreads();

  // ---- stream input row HBM -> f16 LDS via b128 loads (coalesced) ----
  {
    const float4* x4 = (const float4*)(x + (size_t)row * (T_IN * 9));
    const int NV = (T_IN * 9) / 4;                   // 9216, no tail
    for (int i = tid; i < NV; i += NTHREADS) {
      if (i + NTHREADS * 4 < NV) __builtin_prefetch(&x4[i + NTHREADS * 4], 0, 0);
      float4 d = x4[i];
      float dv[4] = {d.x, d.y, d.z, d.w};
      int e = i * 4;
#pragma unroll
      for (int k = 0; k < 4; ++k) {
        int t = (e + k) / 9, f = (e + k) - t * 9;
        bufA[(t + 2) * S1_STRIDE + f] = (_Float16)dv[k];   // +2: left zero halo
      }
    }
  }
  __syncthreads();

  // ================= stage 1: conv(9->16,k3,p2)+relu, pool2 =================
  // K layout kk = k*10 + f  ->  LDS addr = o*10 + kk (contiguous per lane)
  {
    v16h a1;
    v8f cb1;
#pragma unroll
    for (int v = 0; v < 8; ++v) {
      int kb = a_kbase(v, hi);
#pragma unroll
      for (int e = 0; e < 2; ++e) {
        int kk = kb + e;
        float w = 0.f;
        if (kk < 30) { int k = kk / 10, f = kk - k * 10;
                       if (f < 9) w = w1[(n * 9 + f) * 3 + k]; }
        a1[2 * v + e] = (_Float16)w;
      }
      cb1[v] = b1[hi * 8 + v];            // bias folded into WMMA C operand
    }
    for (int tile = wave; tile < 257; tile += NWAVES) {   // out len 4098
      int o = tile * 16 + n;
      v16h b = ld16h_a4(bufA + o * S1_STRIDE + hi * 16);
      v8f c = wmma_f16(a1, b, cb1);
      relu_pool_store(c, h1p, 16, o >> 1, 2049, hi * 8, odd, DUMP_H1);
    }
  }
  __syncthreads();

  // ---- retire input buffer, zero h2p extent (aliased over bufA) ----
  {
    unsigned int* z = (unsigned int*)bufA;
    for (int i = tid; i < H2P_EXT_HALF / 2; i += NTHREADS) z[i] = 0u;
  }
  __syncthreads();

  // ================ stage 2: conv(16->32,k3,p2)+relu, pool2 ================
  // K layout kk = k*16 + c -> addr = o*16 + kk ; K=48 -> 2 splits, 2 M-tiles
  {
    v16h a2[2][2];
    v8f cb2[2];
#pragma unroll
    for (int mt = 0; mt < 2; ++mt) {
#pragma unroll
      for (int s = 0; s < 2; ++s)
#pragma unroll
        for (int v = 0; v < 8; ++v)
#pragma unroll
          for (int e = 0; e < 2; ++e) {
            int kk = s * 32 + a_kbase(v, hi) + e;
            float w = 0.f;
            if (kk < 48) { int k = kk >> 4, c = kk & 15;
                           w = w2[((mt * 16 + n) * 16 + c) * 3 + k]; }
            a2[mt][s][2 * v + e] = (_Float16)w;
          }
#pragma unroll
      for (int v = 0; v < 8; ++v) cb2[mt][v] = b2[mt * 16 + hi * 8 + v];
    }
    for (int tile = wave; tile < 129; tile += NWAVES) {   // out len 2051
      int o = tile * 16 + n;
      const _Float16* hb = h1p + o * 16 + hi * 16;        // 32B aligned
      v16h bA = *(const v16h*)hb;
      v16h bB = *(const v16h*)(hb + 32);
      int p = o >> 1;
#pragma unroll
      for (int mt = 0; mt < 2; ++mt) {
        v8f c = wmma_f16(a2[mt][0], bA, cb2[mt]);
        c = wmma_f16(a2[mt][1], bB, c);
        relu_pool_store(c, bufA, 32, p, 1025, mt * 16 + hi * 8, odd, DUMP_H2);
      }
    }
  }
  __syncthreads();

  // ============ stage 3: conv(32->16,k3,p2)+relu, pool2, mean ============
  // K layout kk = k*32 + c -> addr = o*32 + kk ; K=96 -> 3 splits
  {
    v16h a3[3];
    v8f cb3;
    float acc[8];
#pragma unroll
    for (int sp = 0; sp < 3; ++sp)
#pragma unroll
      for (int v = 0; v < 8; ++v)
#pragma unroll
        for (int e = 0; e < 2; ++e) {
          int kk = sp * 32 + a_kbase(v, hi) + e;          // 0..95, all valid
          int k = kk >> 5, c = kk & 31;
          a3[sp][2 * v + e] = (_Float16)w3[(n * 32 + c) * 3 + k];
        }
#pragma unroll
    for (int v = 0; v < 8; ++v) { cb3[v] = b3[hi * 8 + v]; acc[v] = 0.f; }

    for (int tile = wave; tile < 65; tile += NWAVES) {    // out len 1027
      int o = tile * 16 + n;
      const _Float16* hb = bufA + o * 32 + hi * 16;       // 32B aligned
      v8f c = wmma_f16(a3[0], *(const v16h*)(hb), cb3);
      c = wmma_f16(a3[1], *(const v16h*)(hb + 32), c);
      c = wmma_f16(a3[2], *(const v16h*)(hb + 64), c);
      bool ok = ((o >> 1) < 513);
#pragma unroll
      for (int v = 0; v < 8; ++v) {
        float y = max_xor1_f32(fmaxf(c[v], 0.f));          // both lanes get max
        acc[v] += ok ? y : 0.f;                            // pair double-counts
      }
    }
    // butterfly-sum over the 16 lanes of each half (pairs double-count -> /2)
#pragma unroll
    for (int v = 0; v < 8; ++v) {
      float s = acc[v];
      s += __shfl_xor(s, 1);
      s += __shfl_xor(s, 2);
      s += __shfl_xor(s, 4);
      s += __shfl_xor(s, 8);
      if (n == 0) atomicAdd(&feat_acc[hi * 8 + v], s);
    }
  }
  __syncthreads();

  if (tid < 16)
    feat[(size_t)row * 16 + tid] = feat_acc[tid] * (1.0f / 1026.0f);
}

// ---- feat [256,128] -> MaxPool1d(k=3,s=3) -> out [256,42] ----
__global__ void fd_final_pool(const float* __restrict__ feat,
                              float* __restrict__ out, int n) {
  int i = blockIdx.x * blockDim.x + threadIdx.x;
  if (i >= n) return;
  int b = i / 42, q = i - b * 42;
  const float* r = feat + b * 128 + q * 3;
  out[i] = fmaxf(fmaxf(r[0], r[1]), r[2]);
}

extern "C" void kernel_launch(void* const* d_in, const int* in_sizes, int n_in,
                              void* d_out, int out_size, void* d_ws, size_t ws_size,
                              hipStream_t stream) {
  (void)n_in; (void)ws_size;
  const float* x  = (const float*)d_in[0];
  const float* w1 = (const float*)d_in[1];
  const float* b1 = (const float*)d_in[2];
  const float* w2 = (const float*)d_in[3];
  const float* b2 = (const float*)d_in[4];
  const float* w3 = (const float*)d_in[5];
  const float* b3 = (const float*)d_in[6];
  float* feat = (float*)d_ws;            // [rows, 16] = [256, 128] flat, 128 KB
  float* out  = (float*)d_out;           // [256, 42]

  int rows = in_sizes[0] / (T_IN * 9);   // 2048
  fd_cnn_rows<<<rows, NTHREADS, 0, stream>>>(x, w1, b1, w2, b2, w3, b3, feat);
  int n2 = out_size;                     // 10752
  fd_final_pool<<<(n2 + 255) / 256, 256, 0, stream>>>(feat, out, n2);
}
